// FullFixedTimeCausalConstructiveAttention_32787780338347
// MI455X (gfx1250) — matrix operand order repaired and checked by reading, not compiled, not run on hardware
//
#include <hip/hip_runtime.h>
#include <math.h>

typedef __attribute__((ext_vector_type(16))) _Float16 v16h;
typedef __attribute__((ext_vector_type(8)))  float    v8f;

#define WMMA_F16(a, b, c) \
  __builtin_amdgcn_wmma_f32_16x16x32_f16(false, (a), false, (b), (short)0, (c), false, false)

constexpr int Bc = 8, Lc = 1024, Hc = 8, Ec = 64;
constexpr int BLK_M = 128;   // query rows per workgroup (8 waves x 16 rows)
constexpr int TILE_N = 64;   // key/value tile width
constexpr int LDSTR = 72;    // LDS row stride in halves (bank-conflict pad, keeps 16B align)
constexpr int NWAVE = 8;
constexpr int RS = Hc * Ec;  // l-dimension row stride in elements = 512
constexpr float SCALE = 0.125f;  // 1/sqrt(64), folded into Q fragments

// ---- fragment loaders -------------------------------------------------------
// A-matrix 16x32 f16 layout (ISA 7.12.2): lanes 0-15: m=lane, K groups {0..7,16..23};
// lanes 16-31: m=lane-16, K groups {8..15,24..31}.
__device__ __forceinline__ v16h fragA_global_f32_scaled(const float* rowPtr, int g0) {
  v16h f;
#pragma unroll
  for (int i = 0; i < 8; ++i) f[i] = (_Float16)(rowPtr[g0 + i] * SCALE);
#pragma unroll
  for (int i = 0; i < 8; ++i) f[8 + i] = (_Float16)(rowPtr[g0 + 16 + i] * SCALE);
  return f;
}

__device__ __forceinline__ v16h fragA_lds(const _Float16* tile, int lane, int k0) {
  const int r  = lane & 15;
  const int g0 = (lane < 16) ? 0 : 8;
  const _Float16* p = tile + r * LDSTR + k0 + g0;
  v16h f;
#pragma unroll
  for (int i = 0; i < 8; ++i) f[i] = p[i];
#pragma unroll
  for (int i = 0; i < 8; ++i) f[8 + i] = p[16 + i];
  return f;
}

// B-matrix 32x16 f16 layout (matches A): lanes 0-15: n=lane, K groups {0..7,16..23};
// lanes 16-31: n=lane-16, K groups {8..15,24..31}. tile is [n][k], k contiguous.
__device__ __forceinline__ v16h fragB_lds(const _Float16* tile, int lane, int n0, int k0) {
  const int n  = n0 + (lane & 15);
  const int kk = k0 + ((lane < 16) ? 0 : 8);
  const _Float16* p = tile + n * LDSTR + kk;
  v16h f;
#pragma unroll
  for (int i = 0; i < 8; ++i) f[i] = p[i];
#pragma unroll
  for (int i = 0; i < 8; ++i) f[8 + i] = p[16 + i];
  return f;
}

// ---- per-tile compute body --------------------------------------------------
// EDGE=false: tile is entirely causal-valid (no masking, no diag splice).
// EDGE=true : tile t == mw/64 — apply causal mask and drawn-diagonal splice.
template <bool EDGE>
__device__ __forceinline__ void attend_tile(
    const _Float16* ldsK, const _Float16* ldsVt, _Float16* myP,
    const v16h& qa0, const v16h& qa1,
    v8f (&O)[4], float (&m_i)[8], float (&l_i)[8],
    const float (&dval)[8], int lane, int mlb, int ncol,
    int s0, int mw, int hist)
{
  // ---- S = Q K^T (pre-scaled, 8 WMMAs) ----
  v8f S[4];
#pragma unroll
  for (int nt = 0; nt < 4; ++nt)
#pragma unroll
    for (int r = 0; r < 8; ++r) S[nt][r] = 0.f;
#pragma unroll
  for (int nt = 0; nt < 4; ++nt) {
    v16h b0 = fragB_lds(ldsK, lane, nt * 16, 0);
    S[nt] = WMMA_F16(qa0, b0, S[nt]);
    v16h b1 = fragB_lds(ldsK, lane, nt * 16, 32);
    S[nt] = WMMA_F16(qa1, b1, S[nt]);
  }

  // ---- (edge only) diag splice + causal mask; row max ----
  float rmax[8];
#pragma unroll
  for (int r = 0; r < 8; ++r) rmax[r] = -INFINITY;
#pragma unroll
  for (int nt = 0; nt < 4; ++nt) {
    const int s = s0 + nt * 16 + ncol;
#pragma unroll
    for (int r = 0; r < 8; ++r) {
      float val = S[nt][r];
      if (EDGE) {
        const int m = mw + mlb + r;
        if (s == m && m >= hist) val = dval[r];
        if (s > m) val = -INFINITY;
        S[nt][r] = val;
      }
      rmax[r] = fmaxf(rmax[r], val);
    }
  }

  // ---- online softmax update (reductions stay in 16-lane halves) ----
#pragma unroll
  for (int r = 0; r < 8; ++r) {
    float x = rmax[r];
    x = fmaxf(x, __shfl_xor(x, 1, 32));
    x = fmaxf(x, __shfl_xor(x, 2, 32));
    x = fmaxf(x, __shfl_xor(x, 4, 32));
    x = fmaxf(x, __shfl_xor(x, 8, 32));
    const float mnew  = fmaxf(m_i[r], x);
    const float alpha = __expf(m_i[r] - mnew);  // exp(-inf - finite) == 0 exactly
    m_i[r] = mnew;
    l_i[r] *= alpha;
#pragma unroll
    for (int nt = 0; nt < 4; ++nt) O[nt][r] *= alpha;
  }

  // ---- P = exp(S - m), row sums, stash P as f16 in per-wave LDS ----
  float rsum[8];
#pragma unroll
  for (int r = 0; r < 8; ++r) rsum[r] = 0.f;
#pragma unroll
  for (int nt = 0; nt < 4; ++nt) {
#pragma unroll
    for (int r = 0; r < 8; ++r) {
      const float p = __expf(S[nt][r] - m_i[r]);
      rsum[r] += p;
      myP[(mlb + r) * LDSTR + nt * 16 + ncol] = (_Float16)p;
    }
  }
#pragma unroll
  for (int r = 0; r < 8; ++r) {
    float x = rsum[r];
    x += __shfl_xor(x, 1, 32);
    x += __shfl_xor(x, 2, 32);
    x += __shfl_xor(x, 4, 32);
    x += __shfl_xor(x, 8, 32);
    l_i[r] += x;
  }

  // wave-private LDS round-trip: order the ds stores before the ds loads
  __builtin_amdgcn_fence(__ATOMIC_RELEASE, "wavefront");
  __builtin_amdgcn_wave_barrier();
  __builtin_amdgcn_fence(__ATOMIC_ACQUIRE, "wavefront");

  // ---- O += P V (8 WMMAs) ----
  const v16h pa0 = fragA_lds(myP, lane, 0);
  const v16h pa1 = fragA_lds(myP, lane, 32);
#pragma unroll
  for (int nt = 0; nt < 4; ++nt) {
    v16h vb0 = fragB_lds(ldsVt, lane, nt * 16, 0);
    O[nt] = WMMA_F16(pa0, vb0, O[nt]);
    v16h vb1 = fragB_lds(ldsVt, lane, nt * 16, 32);
    O[nt] = WMMA_F16(pa1, vb1, O[nt]);
  }
}

// ---- kernel -----------------------------------------------------------------
__global__ __launch_bounds__(256)
void ffca_flash_wmma(const float* __restrict__ q,  const float* __restrict__ kmat,
                     const float* __restrict__ vmat, const float* __restrict__ qd,
                     const float* __restrict__ kd, const float* __restrict__ vd,
                     const int* __restrict__ histp, float* __restrict__ out) {
  __shared__ _Float16 ldsK [TILE_N * LDSTR];          // K tile  [s][e]  f16
  __shared__ _Float16 ldsVt[TILE_N * LDSTR];          // V tile  [e][s]  f16 (transposed)
  __shared__ _Float16 ldsP [NWAVE * 16 * LDSTR];      // per-wave P scratch [m][s] f16

  const int hist = *histp;

  const int tid  = threadIdx.x;
  const int lane = tid & 31;
  const int wave = tid >> 5;
  const int m0 = blockIdx.x * BLK_M;
  const int b  = blockIdx.y / Hc;
  const int h  = blockIdx.y % Hc;
  const int mw = m0 + wave * 16;           // this wave's 16 query rows start here

  const long base = (long)b * Lc * Hc * Ec + (long)h * Ec;

  const int mlb  = (lane < 16) ? 0 : 8;    // C-layout row base for this lane half
  const int ncol = lane & 15;              // C-layout column within a 16-wide tile

  // --- Q fragments, pre-scaled by 1/sqrt(E) (q_eff: drawn queries for rows >= hist) ---
  const int mrow = mw + (lane & 15);
  const int g0   = (lane < 16) ? 0 : 8;
  const float* qrow = ((mrow >= hist) ? qd : q) + base + (long)mrow * RS;
  const v16h qa0 = fragA_global_f32_scaled(qrow, g0);
  const v16h qa1 = fragA_global_f32_scaled(qrow + 32, g0);

  // --- drawn-diagonal score scale * (qd[m].kd[m]) ---
  float diag_s;
  {
    const int eo = (lane < 16) ? 0 : 32;
    const float* a = qd + base + (long)mrow * RS + eo;
    const float* c = kd + base + (long)mrow * RS + eo;
    float acc = 0.f;
#pragma unroll
    for (int i = 0; i < 32; ++i) acc = fmaf(a[i], c[i], acc);
    acc += __shfl_xor(acc, 16, 32);
    diag_s = acc * SCALE;
  }
  float dval[8];                            // scaled diag score per C-layout row r
#pragma unroll
  for (int r = 0; r < 8; ++r) dval[r] = __shfl(diag_s, mlb + r, 32);

  v8f O[4];
  float m_i[8], l_i[8];
#pragma unroll
  for (int nt = 0; nt < 4; ++nt)
#pragma unroll
    for (int r = 0; r < 8; ++r) O[nt][r] = 0.f;
#pragma unroll
  for (int r = 0; r < 8; ++r) { m_i[r] = -INFINITY; l_i[r] = 0.f; }

  const int lastTile = (m0 + BLK_M - 1) / TILE_N;   // block-wide inclusive causal bound
  const int tEdge    = mw / TILE_N;                 // the ONE tile crossing this wave's diagonal
  _Float16* myP = &ldsP[wave * 16 * LDSTR];

  for (int t = 0; t <= lastTile; ++t) {
    const int s0 = t * TILE_N;

    // ---- cooperative staging: K -> ldsK[s][e], V -> ldsVt[e][s], fp32->f16 ----
    {
      const int row = tid >> 2;            // 0..63 (s within tile)
      const int c0  = (tid & 3) * 16;      // e chunk
      const float* kg = kmat + base + (long)(s0 + row) * RS + c0;
      const float* vg = vmat + base + (long)(s0 + row) * RS + c0;
      float kvl[16], vvl[16];
#pragma unroll
      for (int i = 0; i < 16; ++i) { kvl[i] = kg[i]; vvl[i] = vg[i]; }
      _Float16* kp = &ldsK[row * LDSTR + c0];
#pragma unroll
      for (int i = 0; i < 16; ++i) kp[i] = (_Float16)kvl[i];
#pragma unroll
      for (int i = 0; i < 16; ++i) ldsVt[(c0 + i) * LDSTR + row] = (_Float16)vvl[i];
      if (t < lastTile) {                  // warm L2 for the next tile
        __builtin_prefetch(kg + (long)TILE_N * RS, 0, 3);
        __builtin_prefetch(vg + (long)TILE_N * RS, 0, 3);
      }
    }
    __syncthreads();

    if (t < tEdge) {          // fully causal-valid tile: no masking code at all
      attend_tile<false>(ldsK, ldsVt, myP, qa0, qa1, O, m_i, l_i, dval,
                         lane, mlb, ncol, s0, mw, hist);
    } else if (t == tEdge) {  // the single diagonal-crossing tile
      attend_tile<true>(ldsK, ldsVt, myP, qa0, qa1, O, m_i, l_i, dval,
                        lane, mlb, ncol, s0, mw, hist);
    }
    __syncthreads();          // protect ldsK/ldsVt before next staging pass
  }

  // ---- epilogue: normalize + drawn-value correction for rows >= hist ----
  float dA[8];
#pragma unroll
  for (int r = 0; r < 8; ++r) {
    dA[r] = __expf(dval[r] - m_i[r]) / l_i[r];   // dval already scaled
  }
#pragma unroll
  for (int nt = 0; nt < 4; ++nt) {
#pragma unroll
    for (int r = 0; r < 8; ++r) {
      const int m = mw + mlb + r;
      const int e = nt * 16 + ncol;
      const long off = base + (long)m * RS + e;
      float val = O[nt][r] / l_i[r];
      if (m >= hist) val += dA[r] * (vd[off] - vmat[off]);
      out[off] = val;
    }
  }
}

// ---- host entry -------------------------------------------------------------
extern "C" void kernel_launch(void* const* d_in, const int* in_sizes, int n_in,
                              void* d_out, int out_size, void* d_ws, size_t ws_size,
                              hipStream_t stream) {
  const float* q    = (const float*)d_in[0];
  const float* kmat = (const float*)d_in[1];
  const float* vmat = (const float*)d_in[2];
  const float* qd   = (const float*)d_in[3];
  const float* kd   = (const float*)d_in[4];
  const float* vd   = (const float*)d_in[5];
  // d_in[6] = attn_mask (unused: causal triu(k=1) implemented analytically)
  const int* histp  = (const int*)d_in[7];
  float* out = (float*)d_out;

  dim3 grid(Lc / BLK_M, Bc * Hc);
  ffca_flash_wmma<<<grid, 256, 0, stream>>>(q, kmat, vmat, qd, kd, vd, histp, out);

  (void)in_sizes; (void)n_in; (void)out_size; (void)d_ws; (void)ws_size;
}
